// LSTMNet_4870492913812
// MI455X (gfx1250) — matrix operand order repaired and checked by reading, not compiled, not run on hardware
//
#include <hip/hip_runtime.h>

// ---------------------------------------------------------------------------
// LSTMNet on gfx1250: bf16 WMMA GEMMs + persistent per-batch-tile LSTM scan
// with TDM (tensor_load_to_lds) double-buffered gate prefetch.
//   B=256, T=256, Din=128, Hemb=H=512, L=3, Dout=1
// ---------------------------------------------------------------------------

typedef __attribute__((ext_vector_type(16))) __bf16 v16bf;
typedef __attribute__((ext_vector_type(8)))  __bf16 v8bf;
typedef __attribute__((ext_vector_type(8)))  float  v8f;
typedef __attribute__((ext_vector_type(4)))  unsigned int uint4v;
typedef __attribute__((ext_vector_type(8)))  int    int8v;
typedef __attribute__((ext_vector_type(4)))  int    int4v;

#define B_SZ   256
#define T_SZ   256
#define DIN    128
#define HID    512     // Hemb == H
#define G4     2048    // 4*H
#define NLAYER 3

__device__ __forceinline__ float fsig(float x) { return 1.0f / (1.0f + __expf(-x)); }
__device__ __forceinline__ float ftanh(float x){ return 2.0f / (1.0f + __expf(-2.0f * x)) - 1.0f; }

__device__ __forceinline__ v8f zero8() {
    v8f z;
#pragma unroll
    for (int r = 0; r < 8; ++r) z[r] = 0.0f;
    return z;
}

__device__ __forceinline__ v16bf join16(v8bf lo, v8bf hi) {
    v16bf v;
#pragma unroll
    for (int i = 0; i < 8; ++i) { v[i] = lo[i]; v[8 + i] = hi[i]; }
    return v;
}

// ---------------------------------------------------------------------------
// TDM: async-copy one [16 x 4H] f32 tile (contiguous 128KB) from global to LDS.
// D# layout per CDNA5 ISA ch.8: group0 = count/lds/global addr, group1 = dims.
// 6-arg builtin flavor (clang-23): (g0, g1, g2, g3, g4, cpol).
// ---------------------------------------------------------------------------
__device__ __forceinline__ void tdm_prefetch_gx(const float* gsrc, unsigned lds_off) {
    unsigned long long ga = (unsigned long long)(uintptr_t)gsrc;
    uint4v g0;
    g0[0] = 1u;                                          // count=1, user descriptor
    g0[1] = lds_off;                                     // lds_addr (bytes)
    g0[2] = (unsigned)(ga & 0xFFFFFFFFu);                // global_addr[31:0]
    g0[3] = (unsigned)((ga >> 32) & 0x1FFFFFFu)          // global_addr[56:32]
          | (2u << 30);                                  // type = 2 ("image")
    int8v g1;
    g1[0] = 0x00020000;                                  // data_size = 2 (4B)
    g1[1] = (int)((G4 & 0xFFFF) << 16);                  // tensor_dim0[15:0]
    g1[2] = (int)(((G4 >> 16) & 0xFFFF)                  // tensor_dim0[31:16]
          | (((B_SZ * T_SZ) & 0xFFFF) << 16));           // tensor_dim1[15:0]
    g1[3] = (int)((((B_SZ * T_SZ) >> 16) & 0xFFFF)       // tensor_dim1[31:16]
          | ((G4 & 0xFFFF) << 16));                      // tile_dim0 = 4H
    g1[4] = 16;                                          // tile_dim1 = 16 rows
    g1[5] = G4;                                          // tensor_dim0_stride[31:0]
    g1[6] = 0;
    g1[7] = 0;
    int4v z4 = {0, 0, 0, 0};
    int8v z8 = {0, 0, 0, 0, 0, 0, 0, 0};
    __builtin_amdgcn_tensor_load_to_lds(g0, g1, z4, z4, z8, 0);
}

// ---------------------------------------------------------------------------
// f32 -> bf16 conversion
// ---------------------------------------------------------------------------
__global__ void cvt_f32_bf16(const float* __restrict__ in, __bf16* __restrict__ out, int n) {
    int i = blockIdx.x * blockDim.x + threadIdx.x;
    if (i < n) out[i] = (__bf16)in[i];
}

// ---------------------------------------------------------------------------
// C[M,N] = A[M,K] @ W[N,K]^T + bias (+bias2).  A,W bf16 row-major, f32 acc.
// Block = 8 waves as 4(M) x 2(N); wave tile = 32(M) x 64(N) so each B
// fragment feeds two WMMAs (1.5 b128 loads per WMMA). WG tile = 128 x 128.
// ---------------------------------------------------------------------------
template <bool OUT_F32>
__global__ void __launch_bounds__(256)
wmma_gemm(const __bf16* __restrict__ A, const __bf16* __restrict__ W,
          const float* __restrict__ bias, const float* __restrict__ bias2,
          float* __restrict__ outF, __bf16* __restrict__ outB,
          int M, int N, int K, int Trows)
{
    const int wave = threadIdx.x >> 5;
    const int lane = threadIdx.x & 31;
    const int l16  = lane & 15;
    const int lh   = lane >> 4;

    const int mbase = blockIdx.y * 128 + (wave >> 1) * 32;
    const int nbase = blockIdx.x * 128 + (wave & 1) * 64;

    v8f acc[2][4];
#pragma unroll
    for (int h = 0; h < 2; ++h)
#pragma unroll
        for (int j = 0; j < 4; ++j) acc[h][j] = zero8();

    const int arow = mbase + l16;

    for (int k = 0; k < K; k += 32) {
        const __bf16* ap0 = A + (size_t)arow * K + k + lh * 8;
        const __bf16* ap1 = ap0 + (size_t)16 * K;
        v16bf a0 = join16(*(const v8bf*)ap0, *(const v8bf*)(ap0 + 16));
        v16bf a1 = join16(*(const v8bf*)ap1, *(const v8bf*)(ap1 + 16));
#pragma unroll
        for (int j = 0; j < 4; ++j) {
            const __bf16* bp = W + (size_t)(nbase + j * 16 + l16) * K + k + lh * 8;
            v16bf bf = join16(*(const v8bf*)bp, *(const v8bf*)(bp + 16));
            acc[0][j] = __builtin_amdgcn_wmma_f32_16x16x32_bf16(
                false, a0, false, bf, (short)0, acc[0][j], false, false);
            acc[1][j] = __builtin_amdgcn_wmma_f32_16x16x32_bf16(
                false, a1, false, bf, (short)0, acc[1][j], false, false);
        }
    }

#pragma unroll
    for (int h = 0; h < 2; ++h) {
#pragma unroll
        for (int j = 0; j < 4; ++j) {
            const int n = nbase + j * 16 + l16;
            float bv = 0.0f;
            if (bias)  bv += bias[n];
            if (bias2) bv += bias2[n];
#pragma unroll
            for (int r = 0; r < 8; ++r) {
                const int mg = mbase + h * 16 + r + lh * 8;   // token row (b*T + t)
                const float v = acc[h][j][r] + bv;
                if (OUT_F32) {
                    size_t idx;
                    if (Trows > 0) {
                        const int bb = mg / Trows, tt = mg - bb * Trows;
                        idx = ((size_t)tt * B_SZ + bb) * N + n;   // [T, B, N]
                    } else {
                        idx = (size_t)mg * N + n;
                    }
                    outF[idx] = v;
                } else {
                    outB[(size_t)mg * N + n] = (__bf16)v;
                }
            }
        }
    }
}

// ---------------------------------------------------------------------------
// Persistent LSTM recurrence. One workgroup per 16 batch rows; h (bf16) in
// LDS, c in registers. Each wave owns 64 columns of *every* gate, so the
// cell update is lane-local. gates_x tiles are TDM-prefetched into a
// double-buffered LDS slab one timestep ahead.
//   gx   : [T, B, 4H] f32   whh : [4H, H] bf16   hout : [B, T, H] bf16
// ---------------------------------------------------------------------------
__global__ void __launch_bounds__(256)
lstm_scan(const float* __restrict__ gx, const __bf16* __restrict__ whh,
          __bf16* __restrict__ hout, int T)
{
    __shared__ __bf16 h_lds[16 * HID];           // 16 KB
    __shared__ float  gxb[2][16 * G4];           // 2 x 128 KB

    const int wave = threadIdx.x >> 5;
    const int lane = threadIdx.x & 31;
    const int l16  = lane & 15;
    const int lh   = lane >> 4;
    const int b0   = blockIdx.x * 16;

    for (int i = threadIdx.x; i < 16 * HID; i += 256) h_lds[i] = (__bf16)0.0f;

    v8f c[4];
#pragma unroll
    for (int j = 0; j < 4; ++j) c[j] = zero8();

    const unsigned lds_gx0 = (unsigned)(uintptr_t)&gxb[0][0];
    const unsigned lds_gx1 = (unsigned)(uintptr_t)&gxb[1][0];

    if (wave == 0)   // prologue: fetch gx tile for t = 0
        tdm_prefetch_gx(gx + (size_t)b0 * G4, lds_gx0);

    __syncthreads();

    for (int t = 0; t < T; ++t) {
        if (wave == 0) {
            __builtin_amdgcn_s_wait_tensorcnt(0);     // gxb[t&1] resident
            if (t + 1 < T)                             // stream next step's tile
                tdm_prefetch_gx(gx + ((size_t)(t + 1) * B_SZ + b0) * G4,
                                ((t + 1) & 1) ? lds_gx1 : lds_gx0);
        }
        __syncthreads();   // gxb[t&1] visible; prior step's LDS reads retired

        // Seed accumulators with the input contribution from LDS.
        const float* gb = gxb[t & 1];
        v8f acc[16];
#pragma unroll
        for (int g = 0; g < 4; ++g) {
#pragma unroll
            for (int j = 0; j < 4; ++j) {
                const int n = g * HID + wave * 64 + j * 16 + l16;
                v8f v;
#pragma unroll
                for (int r = 0; r < 8; ++r) v[r] = gb[(lh * 8 + r) * G4 + n];
                acc[g * 4 + j] = v;
            }
        }

        // gates += h @ w_hh^T  (K = 512 -> 16 steps of 32)
        for (int kk = 0; kk < 16; ++kk) {
            const __bf16* ap = h_lds + l16 * HID + kk * 32 + lh * 8;
            v16bf afrag = join16(*(const v8bf*)ap, *(const v8bf*)(ap + 16));
#pragma unroll
            for (int ti = 0; ti < 16; ++ti) {
                const int g = ti >> 2, j = ti & 3;
                const __bf16* bp = whh +
                    (size_t)(g * HID + wave * 64 + j * 16 + l16) * HID + kk * 32 + lh * 8;
                v16bf bfrag = join16(*(const v8bf*)bp, *(const v8bf*)(bp + 16));
                acc[ti] = __builtin_amdgcn_wmma_f32_16x16x32_bf16(
                    false, afrag, false, bfrag, (short)0, acc[ti], false, false);
            }
        }

        __syncthreads();   // all waves finished reading h_lds for this step

        // Cell update: i,f,g,o for a column sit in acc[j], acc[4+j], acc[8+j], acc[12+j].
#pragma unroll
        for (int j = 0; j < 4; ++j) {
#pragma unroll
            for (int r = 0; r < 8; ++r) {
                const float ii = fsig(acc[j][r]);
                const float ff = fsig(acc[4 + j][r]);
                const float gg = ftanh(acc[8 + j][r]);
                const float oo = fsig(acc[12 + j][r]);
                const float cc = ff * c[j][r] + ii * gg;
                c[j][r] = cc;
                const float hh = oo * ftanh(cc);
                const int m = r + lh * 8;
                const int n = wave * 64 + j * 16 + l16;
                h_lds[m * HID + n] = (__bf16)hh;
                hout[((size_t)(b0 + m) * T + t) * HID + n] = (__bf16)hh;
            }
        }

        __syncthreads();   // h_lds updated for next step
    }
}

// ---------------------------------------------------------------------------
// Decoder: out[b] = dot(h[b, T-1, :], dec_W) + dec_b
// ---------------------------------------------------------------------------
__global__ void decoder_k(const __bf16* __restrict__ hin, const float* __restrict__ dW,
                          const float* __restrict__ db, float* __restrict__ out, int T)
{
    const int b = blockIdx.x * blockDim.x + threadIdx.x;
    if (b >= B_SZ) return;
    const __bf16* hp = hin + ((size_t)b * T + (T - 1)) * HID;
    float s = db[0];
    for (int k = 0; k < HID; ++k) s += (float)hp[k] * dW[k];
    out[b] = s;
}

// ---------------------------------------------------------------------------
extern "C" void kernel_launch(void* const* d_in, const int* in_sizes, int n_in,
                              void* d_out, int out_size, void* d_ws, size_t ws_size,
                              hipStream_t stream) {
    (void)in_sizes; (void)n_in; (void)out_size; (void)ws_size;

    const float* x    = (const float*)d_in[0];   // [B,T,Din]
    const float* embW = (const float*)d_in[1];   // [512,128]
    const float* embB = (const float*)d_in[2];   // [512]
    const float* wih  = (const float*)d_in[3];   // [3,2048,512]
    const float* whh  = (const float*)d_in[4];   // [3,2048,512]
    const float* bih  = (const float*)d_in[5];   // [3,2048]
    const float* bhh  = (const float*)d_in[6];   // [3,2048]
    const float* decW = (const float*)d_in[7];   // [512]
    const float* decb = (const float*)d_in[8];   // [1]
    float* out = (float*)d_out;                  // [256]

    char* ws = (char*)d_ws;
    size_t off = 0;
    auto carve = [&](size_t bytes) -> void* {
        void* p = ws + off;
        off = (off + bytes + 255) & ~(size_t)255;
        return p;
    };

    const size_t MTOK = (size_t)B_SZ * T_SZ;                 // 65536 tokens
    __bf16* xb   = (__bf16*)carve(MTOK * DIN * sizeof(__bf16));
    __bf16* ewb  = (__bf16*)carve((size_t)HID * DIN * sizeof(__bf16));
    __bf16* wihb = (__bf16*)carve((size_t)NLAYER * G4 * HID * sizeof(__bf16));
    __bf16* whhb = (__bf16*)carve((size_t)NLAYER * G4 * HID * sizeof(__bf16));
    __bf16* hA   = (__bf16*)carve(MTOK * HID * sizeof(__bf16));
    __bf16* hB   = (__bf16*)carve(MTOK * HID * sizeof(__bf16));
    float*  gxbf = (float*) carve(MTOK * G4 * sizeof(float));

    // ---- precision conversion ----
    {
        int n;
        n = (int)(MTOK * DIN);
        cvt_f32_bf16<<<(n + 255) / 256, 256, 0, stream>>>(x, xb, n);
        n = HID * DIN;
        cvt_f32_bf16<<<(n + 255) / 256, 256, 0, stream>>>(embW, ewb, n);
        n = NLAYER * G4 * HID;
        cvt_f32_bf16<<<(n + 255) / 256, 256, 0, stream>>>(wih, wihb, n);
        cvt_f32_bf16<<<(n + 255) / 256, 256, 0, stream>>>(whh, whhb, n);
    }

    // ---- embed: h0 = x @ embW^T + embB   (bf16 out) ----
    {
        dim3 grid(HID / 128, (unsigned)(MTOK / 128));
        wmma_gemm<false><<<grid, 256, 0, stream>>>(
            xb, ewb, embB, nullptr, nullptr, hA,
            (int)MTOK, HID, DIN, 0);
    }

    // ---- layers ----
    __bf16* hin = hA;
    __bf16* hot = hB;
    for (int l = 0; l < NLAYER; ++l) {
        // gates_x = h @ w_ih^T + (b_ih + b_hh), stored [T, B, 4H] f32
        dim3 grid(G4 / 128, (unsigned)(MTOK / 128));
        wmma_gemm<true><<<grid, 256, 0, stream>>>(
            hin, wihb + (size_t)l * G4 * HID,
            bih + (size_t)l * G4, bhh + (size_t)l * G4,
            gxbf, nullptr, (int)MTOK, G4, HID, T_SZ);

        // sequential recurrence, persistent per batch tile
        lstm_scan<<<B_SZ / 16, 256, 0, stream>>>(
            gxbf, whhb + (size_t)l * G4 * HID, hot, T_SZ);

        __bf16* tmp = hin; hin = hot; hot = tmp;
    }

    // ---- decoder on last timestep ----
    decoder_k<<<1, 256, 0, stream>>>(hin, decW, decb, out, T_SZ);
}
